// SLatMeshDecoderV3U_14465449853511
// MI455X (gfx1250) — matrix operand order repaired and checked by reading, not compiled, not run on hardware
//
#include <hip/hip_runtime.h>

// ---------------------------------------------------------------------------
// Types for CDNA5 WMMA (wave32, v_wmma_f32_16x16x32_bf16)
// ---------------------------------------------------------------------------
typedef __bf16 bf16;
typedef __bf16 bf16x8  __attribute__((ext_vector_type(8)));
typedef __bf16 bf16x16 __attribute__((ext_vector_type(16)));
typedef float  f32x8   __attribute__((ext_vector_type(8)));
typedef unsigned int u32x4 __attribute__((ext_vector_type(4)));
typedef int    i32x4   __attribute__((ext_vector_type(4)));
typedef int    i32x8   __attribute__((ext_vector_type(8)));

__device__ __forceinline__ bf16 f2bf(float f) {
  union { float f; unsigned u; } v; v.f = f;
  unsigned r = v.u + 0x7FFFu + ((v.u >> 16) & 1u);
  unsigned short h = (unsigned short)(r >> 16);
  return __builtin_bit_cast(bf16, h);
}

__device__ __forceinline__ float gelu_f(float v) {
  return 0.5f * v * (1.0f + tanhf(0.7978845608028654f * (v + 0.044715f * v * v * v)));
}

// ---------------------------------------------------------------------------
// TDM: stage a 64-row x 32-col bf16 tile of Bt (row stride K elements) to LDS.
// D# per CDNA5 ISA ch.8: group0 = {count/flags, lds_addr, global_addr, type=2},
// group1 = {data_size=2B, tensor_dim0=K, tensor_dim1=big, tile=32x64, stride0=K}.
// Tracked on TENSORcnt; issued by one wave only (EXEC ignored by TDM).
// ---------------------------------------------------------------------------
__device__ __forceinline__ void tdm_load_b_tile(const bf16* gsrc, unsigned ldsOff, int K) {
  unsigned long long ga = (unsigned long long)(uintptr_t)gsrc;
  u32x4 g0;
  g0[0] = 1u;                                              // count=1 (valid), user mode
  g0[1] = ldsOff;                                          // lds_addr (bytes)
  g0[2] = (unsigned)(ga & 0xFFFFFFFFu);                    // global_addr[31:0]
  g0[3] = (unsigned)((ga >> 32) & 0x1FFFFFFu) | (2u << 30); // addr[56:32] | type=2
  i32x8 g1;
  g1[0] = 0x00010000;                                      // data_size=1 -> 2 bytes
  g1[1] = (int)(((unsigned)K & 0xFFFFu) << 16);            // tensor_dim0[15:0]
  g1[2] = (int)(((unsigned)K >> 16) & 0xFFFFu);            // tensor_dim0[31:16], dim1 lo=0
  g1[3] = (int)(((1u << 20) >> 16) | (32u << 16));         // tensor_dim1 hi | tile_dim0=32
  g1[4] = 64;                                              // tile_dim1=64, tile_dim2=0
  g1[5] = K;                                               // tensor_dim0_stride[31:0]
  g1[6] = 0;
  g1[7] = 0;
  i32x4 z4 = {0, 0, 0, 0};
#if __clang_major__ >= 23
  i32x8 z8 = {0, 0, 0, 0, 0, 0, 0, 0};
  __builtin_amdgcn_tensor_load_to_lds(g0, g1, z4, z4, z8, 0);
#else
  __builtin_amdgcn_tensor_load_to_lds(g0, g1, z4, z4, 0);
#endif
}

// ---------------------------------------------------------------------------
// bf16 WMMA GEMM: out[M,N] = act(A[M,K] @ Bt[N,K]^T + bias) (+res)
//  - block (8 waves) computes 128x64 tile; B panel staged to LDS by TDM,
//    double-buffered, shared by all 8 waves; A rows gathered per lane.
//  - rowsIdx: optional gather (neg -> zero row via zrow), then >> rowShift.
//  - epilogue: act 0=none 1=gelu 2=silu; res added after act; fp32/bf16 out.
//  Requires: M % 128 == 0, K % 32 == 0, Bt has >= ceil(N/64)*64 rows.
// ---------------------------------------------------------------------------
__global__ __launch_bounds__(256) void gemm_bf16_wmma(
    const bf16* __restrict__ A, const bf16* __restrict__ Bt,
    const float* __restrict__ bias, const float* __restrict__ res,
    float* __restrict__ outF, bf16* __restrict__ outH,
    const int* __restrict__ rowsIdx, int rowShift,
    const bf16* __restrict__ zrow,
    int M, int N, int K, int act)
{
  __shared__ bf16 smB[2][64 * 32];   // two 4KB B panels (64 cols x 32 k)
  int tilesN = (N + 63) >> 6;
  int bidRow = blockIdx.x / tilesN;
  int tn     = blockIdx.x - bidRow * tilesN;
  int wave   = threadIdx.x >> 5;
  int lane   = threadIdx.x & 31;
  int tm     = bidRow * 8 + wave;          // 16-row tile index
  int l16    = lane & 15;
  int half   = lane >> 4;

  // Per-lane A row (lane half selects contiguous K run {0..15}
  // vs {16..31} of each 16x32 tile, matching the B fragment)
  int gr  = tm * 16 + l16;
  int src = rowsIdx ? rowsIdx[gr] : gr;
  const bf16* Ar = (src >= 0) ? (A + (long long)(src >> rowShift) * K) : zrow;

  f32x8 acc0 = {0,0,0,0,0,0,0,0};
  f32x8 acc1 = acc0, acc2 = acc0, acc3 = acc0;

  const bf16* Bg = Bt + (long long)(tn * 64) * K;  // global base of this B panel
  unsigned lds0 = (unsigned)(uintptr_t)&smB[0][0];
  unsigned lds1 = (unsigned)(uintptr_t)&smB[1][0];

  if (wave == 0) {
    tdm_load_b_tile(Bg, lds0, K);
    __builtin_amdgcn_s_wait_tensorcnt(0);
  }
  __syncthreads();

  int nchunks = K >> 5;
  int cur = 0;
  for (int ci = 0; ci < nchunks; ++ci) {
    int k0 = ci << 5;
    if (wave == 0 && ci + 1 < nchunks)
      tdm_load_b_tile(Bg + k0 + 32, cur ? lds0 : lds1, K);

    __builtin_prefetch(Ar + k0 + 128, 0, 0);
    bf16x8 a0 = *(const bf16x8*)(Ar + k0 + half * 16);
    bf16x8 a1 = *(const bf16x8*)(Ar + k0 + half * 16 + 8);
    bf16x16 afrag = __builtin_shufflevector(a0, a1,
        0,1,2,3,4,5,6,7,8,9,10,11,12,13,14,15);

    // B fragments from LDS: sub-tile s covers panel rows s*16..s*16+15
    const bf16* sb = &smB[cur][0];
    bf16x16 b0 = *(const bf16x16*)(sb + (0 * 16 + l16) * 32 + half * 16);
    bf16x16 b1 = *(const bf16x16*)(sb + (1 * 16 + l16) * 32 + half * 16);
    bf16x16 b2 = *(const bf16x16*)(sb + (2 * 16 + l16) * 32 + half * 16);
    bf16x16 b3 = *(const bf16x16*)(sb + (3 * 16 + l16) * 32 + half * 16);

    acc0 = __builtin_amdgcn_wmma_f32_16x16x32_bf16(false, afrag, false, b0, (short)0, acc0, false, false);
    acc1 = __builtin_amdgcn_wmma_f32_16x16x32_bf16(false, afrag, false, b1, (short)0, acc1, false, false);
    acc2 = __builtin_amdgcn_wmma_f32_16x16x32_bf16(false, afrag, false, b2, (short)0, acc2, false, false);
    acc3 = __builtin_amdgcn_wmma_f32_16x16x32_bf16(false, afrag, false, b3, (short)0, acc3, false, false);

    if (wave == 0 && ci + 1 < nchunks)
      __builtin_amdgcn_s_wait_tensorcnt(0);
    __syncthreads();
    cur ^= 1;
  }

  f32x8 accs[4] = {acc0, acc1, acc2, acc3};
  #pragma unroll
  for (int s = 0; s < 4; ++s) {
    int col = tn * 64 + s * 16 + l16;
    if (col >= N) continue;
    float bv = bias ? bias[col] : 0.0f;
    #pragma unroll
    for (int r = 0; r < 8; ++r) {
      int m = tm * 16 + half * 8 + r;  // ISA C/D layout: VGPR r -> M=r / M=8+r
      long long idx = (long long)m * N + col;
      float v = accs[s][r] + bv;
      if (act == 1)      v = gelu_f(v);
      else if (act == 2) v = v / (1.0f + __expf(-v));
      if (res)  v += res[idx];
      if (outF) outF[idx] = v;
      if (outH) outH[idx] = f2bf(v);
    }
  }
}

// ---------------------------------------------------------------------------
// Weight prep: fp32 [R, Nc] -> bf16 transposed [Npad, R], zero-padded rows
// ---------------------------------------------------------------------------
__global__ void k_transpose_bf16(const float* __restrict__ src, bf16* __restrict__ dst,
                                 int R, int Nc, int Npad) {
  long long i = (long long)blockIdx.x * blockDim.x + threadIdx.x;
  long long total = (long long)Npad * R;
  if (i >= total) return;
  int r = (int)(i % R);
  int n = (int)(i / R);
  dst[i] = (n < Nc) ? f2bf(src[(long long)r * Nc + n]) : f2bf(0.0f);
}

__global__ void k_f32_to_bf16(const float* __restrict__ src, bf16* __restrict__ dst, long long total) {
  long long i = (long long)blockIdx.x * blockDim.x + threadIdx.x;
  if (i < total) dst[i] = f2bf(src[i]);
}

// ---------------------------------------------------------------------------
// Input embedding: h = feats @ in_W + in_b + posemb(coords)
// ---------------------------------------------------------------------------
__global__ void k_embed(const float* __restrict__ feats, const int* __restrict__ coords,
                        const float* __restrict__ inW, const float* __restrict__ inb,
                        float* __restrict__ h, int Nrows, int C) {
  long long i = (long long)blockIdx.x * blockDim.x + threadIdx.x;
  if (i >= (long long)Nrows * C) return;
  int row = (int)(i / C), c = (int)(i % C);
  float acc = inb[c];
  #pragma unroll
  for (int k = 0; k < 8; ++k) acc += feats[row * 8 + k] * inW[k * C + c];
  int axis = c / 256, j = c % 256;
  float coord = (float)coords[row * 4 + 1 + axis];
  float f = __powf(10000.0f, -((float)(j & 127)) / 128.0f);
  float a = coord * f;
  acc += (j < 128) ? __sinf(a) : __cosf(a);
  h[i] = acc;
}

// ---------------------------------------------------------------------------
// Stable window ordering (counting sort): 512 windows, single block
// ---------------------------------------------------------------------------
__global__ __launch_bounds__(512) void k_order(const int* __restrict__ coords, int Nrows,
                                               int shift, int* __restrict__ order) {
  int w = threadIdx.x;  // 0..511
  __shared__ int cnt[512];
  int c = 0;
  for (int i = 0; i < Nrows; ++i) {
    int wx = ((coords[i * 4 + 1] + shift) & 63) >> 3;
    int wy = ((coords[i * 4 + 2] + shift) & 63) >> 3;
    int wz = ((coords[i * 4 + 3] + shift) & 63) >> 3;
    int wid = (wx * 8 + wy) * 8 + wz;
    if (wid == w) c++;
  }
  cnt[w] = c;
  __syncthreads();
  int off = 0;
  for (int u = 0; u < w; ++u) off += cnt[u];
  int p = off;
  for (int i = 0; i < Nrows; ++i) {
    int wx = ((coords[i * 4 + 1] + shift) & 63) >> 3;
    int wy = ((coords[i * 4 + 2] + shift) & 63) >> 3;
    int wz = ((coords[i * 4 + 3] + shift) & 63) >> 3;
    int wid = (wx * 8 + wy) * 8 + wz;
    if (wid == w) order[p++] = i;
  }
}

// ---------------------------------------------------------------------------
// LayerNorm -> bf16
// ---------------------------------------------------------------------------
__global__ __launch_bounds__(256) void k_ln_bf16(const float* __restrict__ x,
                                                 const float* __restrict__ g,
                                                 const float* __restrict__ b,
                                                 bf16* __restrict__ out, int C) {
  int row = blockIdx.x;
  const float* xr = x + (long long)row * C;
  __shared__ float s1[256], s2[256];
  float a = 0.0f, a2 = 0.0f;
  for (int c = threadIdx.x; c < C; c += 256) { float v = xr[c]; a += v; a2 += v * v; }
  s1[threadIdx.x] = a; s2[threadIdx.x] = a2;
  __syncthreads();
  for (int st = 128; st > 0; st >>= 1) {
    if (threadIdx.x < st) { s1[threadIdx.x] += s1[threadIdx.x + st]; s2[threadIdx.x] += s2[threadIdx.x + st]; }
    __syncthreads();
  }
  float mean = s1[0] / C;
  float var  = s2[0] / C - mean * mean;
  float inv  = rsqrtf(var + 1e-5f);
  for (int c = threadIdx.x; c < C; c += 256)
    out[(long long)row * C + c] = f2bf((xr[c] - mean) * inv * g[c] + b[c]);
}

// ---------------------------------------------------------------------------
// Windowed attention (16 tokens/window, 12 heads, DH=64); writes bf16 to token rows
// ---------------------------------------------------------------------------
__global__ __launch_bounds__(64) void k_attention(const float* __restrict__ qkv,
                                                  const int* __restrict__ order,
                                                  bf16* __restrict__ o, int nw) {
  long long t = (long long)blockIdx.x * blockDim.x + threadIdx.x;
  if (t >= (long long)nw * 12 * 16) return;
  int q  = (int)(t % 16);
  int hh = (int)((t / 16) % 12);
  int w  = (int)(t / 192);
  int tokq = order[w * 16 + q];
  const float* qv = qkv + (long long)tokq * 2304 + hh * 64;
  float qr[64];
  #pragma unroll
  for (int d = 0; d < 64; ++d) qr[d] = qv[d];
  float s[16], mx = -1e30f;
  for (int p = 0; p < 16; ++p) {
    int tok = order[w * 16 + p];
    const float* kv = qkv + (long long)tok * 2304 + 768 + hh * 64;
    float dot = 0.0f;
    #pragma unroll
    for (int d = 0; d < 64; ++d) dot += qr[d] * kv[d];
    s[p] = dot * 0.125f;
    mx = fmaxf(mx, s[p]);
  }
  float den = 0.0f;
  for (int p = 0; p < 16; ++p) { s[p] = __expf(s[p] - mx); den += s[p]; }
  float accv[64];
  #pragma unroll
  for (int d = 0; d < 64; ++d) accv[d] = 0.0f;
  for (int p = 0; p < 16; ++p) {
    int tok = order[w * 16 + p];
    const float* vv = qkv + (long long)tok * 2304 + 1536 + hh * 64;
    float a = s[p] / den;
    #pragma unroll
    for (int d = 0; d < 64; ++d) accv[d] += a * vv[d];
  }
  bf16* orow = o + (long long)tokq * 768 + hh * 64;
  #pragma unroll
  for (int d = 0; d < 64; ++d) orow[d] = f2bf(accv[d]);
}

// ---------------------------------------------------------------------------
// GroupNorm over whole tensor (G=32): deterministic two-stage reduction
// ---------------------------------------------------------------------------
#define GN_BLOCKS 240
__global__ __launch_bounds__(256) void k_gn_partial(const float* __restrict__ x, int Nrows,
                                                    int C, float* __restrict__ part) {
  int g = threadIdx.x >> 3, sub = threadIdx.x & 7;
  int cg = C / 32;
  float s = 0.0f, s2 = 0.0f;
  for (int row = blockIdx.x; row < Nrows; row += gridDim.x) {
    const float* xr = x + (long long)row * C + g * cg;
    for (int c = sub; c < cg; c += 8) { float v = xr[c]; s += v; s2 += v * v; }
  }
  __shared__ float sh[512];
  sh[threadIdx.x * 2] = s; sh[threadIdx.x * 2 + 1] = s2;
  __syncthreads();
  if (sub == 0) {
    float S = 0.0f, S2 = 0.0f;
    for (int u = 0; u < 8; ++u) { S += sh[(g * 8 + u) * 2]; S2 += sh[(g * 8 + u) * 2 + 1]; }
    part[((long long)blockIdx.x * 32 + g) * 2]     = S;
    part[((long long)blockIdx.x * 32 + g) * 2 + 1] = S2;
  }
}

__global__ void k_gn_final(const float* __restrict__ part, int nblk, float cnt,
                           float* __restrict__ stats) {
  int g = threadIdx.x;
  if (g >= 32) return;
  float S = 0.0f, S2 = 0.0f;
  for (int b = 0; b < nblk; ++b) { S += part[(b * 32 + g) * 2]; S2 += part[(b * 32 + g) * 2 + 1]; }
  float m = S / cnt;
  float var = S2 / cnt - m * m;
  stats[g * 2] = m;
  stats[g * 2 + 1] = rsqrtf(var + 1e-5f);
}

__global__ void k_gn_silu(const float* __restrict__ x, const float* __restrict__ stats,
                          const float* __restrict__ g, const float* __restrict__ b,
                          bf16* __restrict__ out, long long total, int C) {
  long long i = (long long)blockIdx.x * blockDim.x + threadIdx.x;
  if (i >= total) return;
  int c = (int)(i % C);
  int grp = c / (C / 32);
  float v = (x[i] - stats[grp * 2]) * stats[grp * 2 + 1] * g[c] + b[c];
  out[i] = f2bf(v / (1.0f + __expf(-v)));
}

// ---------------------------------------------------------------------------
// Sparse-grid helpers
// ---------------------------------------------------------------------------
__global__ void k_subdiv_coords(const int* __restrict__ src, int stride, int offc,
                                int* __restrict__ dst, int Nin) {
  long long t = (long long)blockIdx.x * blockDim.x + threadIdx.x;
  if (t >= (long long)Nin * 8) return;
  int i = (int)(t >> 3), s = (int)(t & 7);
  dst[t * 3 + 0] = src[(long long)i * stride + offc + 0] * 2 + ((s >> 2) & 1);
  dst[t * 3 + 1] = src[(long long)i * stride + offc + 1] * 2 + ((s >> 1) & 1);
  dst[t * 3 + 2] = src[(long long)i * stride + offc + 2] * 2 + (s & 1);
}

__global__ void k_fill_i32(int* __restrict__ p, long long total, int v) {
  long long i = (long long)blockIdx.x * blockDim.x + threadIdx.x;
  if (i < total) p[i] = v;
}

__global__ void k_set_grid(const int* __restrict__ c, int* __restrict__ grid, int M, int res) {
  long long j = (long long)blockIdx.x * blockDim.x + threadIdx.x;
  if (j >= M) return;
  long long lin = ((long long)c[j * 3] * res + c[j * 3 + 1]) * res + c[j * 3 + 2];
  grid[lin] = (int)j;
}

__global__ void k_nbr(const int* __restrict__ c, const int* __restrict__ grid,
                      int M, int res, int* __restrict__ nbr) {
  long long t = (long long)blockIdx.x * blockDim.x + threadIdx.x;
  if (t >= 27LL * M) return;
  int j = (int)(t % M), k = (int)(t / M);
  int x = c[j * 3 + 0] + (k / 9) - 1;
  int y = c[j * 3 + 1] + ((k / 3) % 3) - 1;
  int z = c[j * 3 + 2] + (k % 3) - 1;
  int v = -1;
  if ((unsigned)x < (unsigned)res && (unsigned)y < (unsigned)res && (unsigned)z < (unsigned)res)
    v = grid[((long long)x * res + y) * res + z];
  nbr[t] = v;
}

// ---------------------------------------------------------------------------
// Fused occupancy head: occ = gelu(h @ W1 + b1) @ W2 + b2 (one block / point)
// ---------------------------------------------------------------------------
__global__ void k_occ(const float* __restrict__ h, const float* __restrict__ W1,
                      const float* __restrict__ b1, const float* __restrict__ W2,
                      const float* __restrict__ b2, float* __restrict__ occ,
                      int Cc, int HID) {
  int j = blockIdx.x;
  __shared__ float sh[192];
  __shared__ float red[1024];
  const float* hr = h + (long long)j * Cc;
  for (int c = threadIdx.x; c < Cc; c += blockDim.x) sh[c] = hr[c];
  __syncthreads();
  int u = threadIdx.x;
  float acc = b1[u];
  for (int c = 0; c < Cc; ++c) acc += sh[c] * W1[c * HID + u];
  float gv = gelu_f(acc) * W2[u];
  red[u] = gv;
  for (int z = u + HID; z < 1024; z += HID) red[z] = 0.0f;
  __syncthreads();
  for (int st = 512; st > 0; st >>= 1) {
    if (u < st) red[u] += red[u + st];
    __syncthreads();
  }
  if (u == 0) occ[j] = red[0] + b2[0];
}

// ---------------------------------------------------------------------------
// Host orchestration
// ---------------------------------------------------------------------------
static inline int cdiv(long long a, long long b) { return (int)((a + b - 1) / b); }

static void launch_gemm(const bf16* A, const bf16* Bt, const float* bias, const float* res,
                        float* outF, bf16* outH, const int* rowsIdx, int rowShift,
                        const bf16* zrow, long long M, int N, int K, int act,
                        hipStream_t s) {
  int tilesN = (N + 63) >> 6;
  int blocks = (int)((M / 128) * tilesN);
  gemm_bf16_wmma<<<blocks, 256, 0, s>>>(A, Bt, bias, res, outF, outH, rowsIdx, rowShift,
                                        zrow, (int)M, N, K, act);
}

static void launch_T(const float* src, bf16* dst, int R, int Nc, int Npad, hipStream_t s) {
  long long total = (long long)Npad * R;
  k_transpose_bf16<<<cdiv(total, 256), 256, 0, s>>>(src, dst, R, Nc, Npad);
}

extern "C" void kernel_launch(void* const* d_in, const int* in_sizes, int n_in,
                              void* d_out, int out_size, void* d_ws, size_t ws_size,
                              hipStream_t stream) {
  (void)in_sizes; (void)n_in; (void)out_size; (void)ws_size;

  const int N0 = 8192, C = 768;
  const int N1 = 65536, C1 = 192;
  const int N2 = 524288, C2 = 96;
  const long long OCC1_OFF = (long long)N2 * 77;
  const long long OCC2_OFF = OCC1_OFF + N1;

  const float* feats  = (const float*)d_in[0];
  const int*   coords = (const int*)d_in[1];
  const float* in_W   = (const float*)d_in[2];
  const float* in_b   = (const float*)d_in[3];
  auto blk  = [&](int i, int j) { return (const float*)d_in[4 + i * 12 + j]; };
  auto up1p = [&](int j) { return (const float*)d_in[148 + j]; };
  auto up2p = [&](int j) { return (const float*)d_in[162 + j]; };
  const float* out_W = (const float*)d_in[176];
  const float* out_b = (const float*)d_in[177];
  float* dout = (float*)d_out;

  char* ws = (char*)d_ws;
  size_t off = 0;
  auto A = [&](size_t bytes) -> char* {
    char* p = ws + off;
    off = (off + bytes + 255) & ~(size_t)255;
    return p;
  };

  // ---- persistent region ----
  bf16 *qkvT[12], *projT[12], *mlp1T[12], *mlp2T[12];
  for (int i = 0; i < 12; ++i) {
    qkvT[i]  = (bf16*)A(2304LL * 768 * 2);
    projT[i] = (bf16*)A(768LL * 768 * 2);
    mlp1T[i] = (bf16*)A(3072LL * 768 * 2);
    mlp2T[i] = (bf16*)A(768LL * 3072 * 2);
  }
  bf16* c1T1 = (bf16*)A(27LL * 192 * 768 * 2);
  bf16* c2T1 = (bf16*)A(27LL * 192 * 192 * 2);
  bf16* skT1 = (bf16*)A(192LL * 768 * 2);
  bf16* c1T2 = (bf16*)A(27LL * 128 * 192 * 2);
  bf16* c2T2 = (bf16*)A(27LL * 128 * 96 * 2);
  bf16* skT2 = (bf16*)A(128LL * 192 * 2);
  bf16* outT = (bf16*)A(128LL * 96 * 2);
  float* h     = (float*)A((long long)N0 * C * 4);
  float* buf2  = (float*)A((long long)N1 * C1 * 4);
  int* c2c     = (int*)A((long long)N1 * 3 * 4);
  int* order0  = (int*)A(N0 * 4);
  int* order1  = (int*)A(N0 * 4);
  float* gnpart = (float*)A(GN_BLOCKS * 32 * 2 * 4);
  float* gnstat = (float*)A(32 * 2 * 4);
  bf16* zrow    = (bf16*)A(3072LL * 2);   // zero A-row for gather misses
  size_t mark = off;

  // ---- zero row + weight prep: fp32 -> bf16 transposed (+pad) ----
  k_fill_i32<<<cdiv(1536, 256), 256, 0, stream>>>((int*)zrow, 1536, 0);
  for (int i = 0; i < 12; ++i) {
    launch_T(blk(i, 2),  qkvT[i],  768, 2304, 2304, stream);
    launch_T(blk(i, 4),  projT[i], 768, 768, 768, stream);
    launch_T(blk(i, 8),  mlp1T[i], 768, 3072, 3072, stream);
    launch_T(blk(i, 10), mlp2T[i], 3072, 768, 768, stream);
  }
  for (int t = 0; t < 27; ++t) {
    launch_T(up1p(2) + (long long)t * 768 * 192, c1T1 + (long long)t * 192 * 768, 768, 192, 192, stream);
    launch_T(up1p(6) + (long long)t * 192 * 192, c2T1 + (long long)t * 192 * 192, 192, 192, 192, stream);
    launch_T(up2p(2) + (long long)t * 192 * 96,  c1T2 + (long long)t * 128 * 192, 192, 96, 128, stream);
    launch_T(up2p(6) + (long long)t * 96 * 96,   c2T2 + (long long)t * 128 * 96,  96, 96, 128, stream);
  }
  launch_T(up1p(8), skT1, 768, 192, 192, stream);
  launch_T(up2p(8), skT2, 192, 96, 128, stream);
  launch_T(out_W,   outT, 96, 77, 128, stream);

  // ---- orderings + embedding ----
  k_order<<<1, 512, 0, stream>>>(coords, N0, 0, order0);
  k_order<<<1, 512, 0, stream>>>(coords, N0, 4, order1);
  k_embed<<<cdiv((long long)N0 * C, 256), 256, 0, stream>>>(feats, coords, in_W, in_b, h, N0, C);

  // ---- transformer ----
  off = mark;
  bf16*  xln  = (bf16*)A((long long)N0 * C * 2);
  float* qkv  = (float*)A((long long)N0 * 2304 * 4);
  bf16*  atno = (bf16*)A((long long)N0 * C * 2);
  bf16*  mid  = (bf16*)A((long long)N0 * 3072 * 2);
  for (int i = 0; i < 12; ++i) {
    k_ln_bf16<<<N0, 256, 0, stream>>>(h, blk(i, 0), blk(i, 1), xln, C);
    launch_gemm(xln, qkvT[i], blk(i, 3), nullptr, qkv, nullptr, nullptr, 0, zrow, N0, 2304, C, 0, stream);
    k_attention<<<cdiv((long long)512 * 12 * 16, 64), 64, 0, stream>>>(qkv, (i % 2 == 0) ? order0 : order1, atno, 512);
    launch_gemm(atno, projT[i], blk(i, 5), h, h, nullptr, nullptr, 0, zrow, N0, C, C, 0, stream);
    k_ln_bf16<<<N0, 256, 0, stream>>>(h, blk(i, 6), blk(i, 7), xln, C);
    launch_gemm(xln, mlp1T[i], blk(i, 9), nullptr, nullptr, mid, nullptr, 0, zrow, N0, 3072, C, 1, stream);
    launch_gemm(mid, mlp2T[i], blk(i, 11), h, h, nullptr, nullptr, 0, zrow, N0, C, 3072, 0, stream);
  }

  // ---- up1: 8192 -> 65536, 768 -> 192, res 128 ----
  off = mark;
  bf16* a1    = (bf16*)A((long long)N0 * C * 2);
  bf16* hbf   = (bf16*)A((long long)N0 * C * 2);
  int* grid1  = (int*)A(2097152LL * 4);
  int* nbr1   = (int*)A(27LL * N1 * 4);
  float* buf1 = (float*)A((long long)N1 * C1 * 4);
  bf16* h1s   = (bf16*)A((long long)N1 * C1 * 2);

  k_gn_partial<<<GN_BLOCKS, 256, 0, stream>>>(h, N0, C, gnpart);
  k_gn_final<<<1, 32, 0, stream>>>(gnpart, GN_BLOCKS, (float)((long long)N0 * (C / 32)), gnstat);
  k_gn_silu<<<cdiv((long long)N0 * C, 256), 256, 0, stream>>>(h, gnstat, up1p(0), up1p(1), a1, (long long)N0 * C, C);

  k_subdiv_coords<<<cdiv((long long)N1, 256), 256, 0, stream>>>(coords, 4, 1, c2c, N0);
  k_fill_i32<<<cdiv(2097152LL, 256), 256, 0, stream>>>(grid1, 2097152LL, -1);
  k_set_grid<<<cdiv((long long)N1, 256), 256, 0, stream>>>(c2c, grid1, N1, 128);
  k_nbr<<<cdiv(27LL * N1, 256), 256, 0, stream>>>(c2c, grid1, N1, 128, nbr1);

  for (int t = 0; t < 27; ++t)
    launch_gemm(a1, c1T1 + (long long)(26 - t) * 192 * 768, t == 0 ? up1p(3) : nullptr,
                t == 0 ? nullptr : buf1, buf1, nullptr, nbr1 + (long long)t * N1, 3,
                zrow, N1, C1, C, 0, stream);

  k_gn_partial<<<GN_BLOCKS, 256, 0, stream>>>(buf1, N1, C1, gnpart);
  k_gn_final<<<1, 32, 0, stream>>>(gnpart, GN_BLOCKS, (float)((long long)N1 * (C1 / 32)), gnstat);
  k_gn_silu<<<cdiv((long long)N1 * C1, 256), 256, 0, stream>>>(buf1, gnstat, up1p(4), up1p(5), h1s, (long long)N1 * C1, C1);

  for (int t = 0; t < 27; ++t)
    launch_gemm(h1s, c2T1 + (long long)(26 - t) * 192 * 192, t == 0 ? up1p(7) : nullptr,
                t == 0 ? nullptr : buf2, buf2, nullptr, nbr1 + (long long)t * N1, 0,
                zrow, N1, C1, C1, 0, stream);

  k_f32_to_bf16<<<cdiv((long long)N0 * C, 256), 256, 0, stream>>>(h, hbf, (long long)N0 * C);
  launch_gemm(hbf, skT1, up1p(9), buf2, buf2, nullptr, nullptr, 3, zrow, N1, C1, C, 0, stream);

  k_occ<<<N1, 768, 0, stream>>>(buf2, up1p(10), up1p(11), up1p(12), up1p(13), dout + OCC1_OFF, C1, 768);

  // ---- up2: 65536 -> 524288, 192 -> 96, res 256 ----
  off = mark;
  bf16* a2    = (bf16*)A((long long)N1 * C1 * 2);
  bf16* h2bf  = (bf16*)A((long long)N1 * C1 * 2);
  int* c3c    = (int*)A((long long)N2 * 3 * 4);
  int* grid2  = (int*)A(16777216LL * 4);
  int* nbr2   = (int*)A(27LL * N2 * 4);
  float* g1   = (float*)A((long long)N2 * C2 * 4);
  bf16* g1s   = (bf16*)A((long long)N2 * C2 * 2);
  float* g2   = (float*)A((long long)N2 * C2 * 4);
  bf16* g2b   = (bf16*)A((long long)N2 * C2 * 2);

  k_gn_partial<<<GN_BLOCKS, 256, 0, stream>>>(buf2, N1, C1, gnpart);
  k_gn_final<<<1, 32, 0, stream>>>(gnpart, GN_BLOCKS, (float)((long long)N1 * (C1 / 32)), gnstat);
  k_gn_silu<<<cdiv((long long)N1 * C1, 256), 256, 0, stream>>>(buf2, gnstat, up2p(0), up2p(1), a2, (long long)N1 * C1, C1);

  k_subdiv_coords<<<cdiv((long long)N2, 256), 256, 0, stream>>>(c2c, 3, 0, c3c, N1);
  k_fill_i32<<<cdiv(16777216LL, 256), 256, 0, stream>>>(grid2, 16777216LL, -1);
  k_set_grid<<<cdiv((long long)N2, 256), 256, 0, stream>>>(c3c, grid2, N2, 256);
  k_nbr<<<cdiv(27LL * N2, 256), 256, 0, stream>>>(c3c, grid2, N2, 256, nbr2);

  for (int t = 0; t < 27; ++t)
    launch_gemm(a2, c1T2 + (long long)(26 - t) * 128 * 192, t == 0 ? up2p(3) : nullptr,
                t == 0 ? nullptr : g1, g1, nullptr, nbr2 + (long long)t * N2, 3,
                zrow, N2, C2, C1, 0, stream);

  k_gn_partial<<<GN_BLOCKS, 256, 0, stream>>>(g1, N2, C2, gnpart);
  k_gn_final<<<1, 32, 0, stream>>>(gnpart, GN_BLOCKS, (float)((long long)N2 * (C2 / 32)), gnstat);
  k_gn_silu<<<cdiv((long long)N2 * C2, 256), 256, 0, stream>>>(g1, gnstat, up2p(4), up2p(5), g1s, (long long)N2 * C2, C2);

  for (int t = 0; t < 27; ++t)
    launch_gemm(g1s, c2T2 + (long long)(26 - t) * 128 * 96, t == 0 ? up2p(7) : nullptr,
                t == 0 ? nullptr : g2, g2, nullptr, nbr2 + (long long)t * N2, 0,
                zrow, N2, C2, C2, 0, stream);

  k_f32_to_bf16<<<cdiv((long long)N1 * C1, 256), 256, 0, stream>>>(buf2, h2bf, (long long)N1 * C1);
  launch_gemm(h2bf, skT2, up2p(9), g2, g2, nullptr, nullptr, 3, zrow, N2, C2, C1, 0, stream);

  k_occ<<<N2, 384, 0, stream>>>(g2, up2p(10), up2p(11), up2p(12), up2p(13), dout + OCC2_OFF, C2, 384);

  k_f32_to_bf16<<<cdiv((long long)N2 * C2, 256), 256, 0, stream>>>(g2, g2b, (long long)N2 * C2);
  launch_gemm(g2b, outT, out_b, nullptr, dout, nullptr, nullptr, 0, zrow, N2, 77, C2, 0, stream);
}
